// MultiDoubleStreamBlockLoraProcessor_41592463294689
// MI455X (gfx1250) — matrix operand order repaired and checked by reading, not compiled, not run on hardware
//
#include <hip/hip_runtime.h>
#include <hip/hip_bf16.h>

// ---------------------------------------------------------------------------
// MI455X (gfx1250, wave32) implementation of the double-stream LoRA attention
// block. All matmuls go through v_wmma_f32_16x16x32_bf16 (f32 accumulate).
// Big GEMMs are LDS-staged (convert f32->bf16 once, 8-wave reuse) with 32x64
// register blocking per wave (8 WMMAs per K-step per wave).
// ---------------------------------------------------------------------------

#define DIMV   3072
#define HEADSV 24
#define HDV    128
#define LIMG   2048
#define LTXT   512
#define STOT   2560   // LTXT + LIMG
#define RANKV  16
#define NLV    2

typedef __attribute__((ext_vector_type(16))) __bf16 bf16x16;
typedef __attribute__((ext_vector_type(8)))  float  floatx8;

static __device__ inline floatx8 wmma_bf16(bf16x16 a, bf16x16 b, floatx8 c) {
  // (neg_a, A, neg_b, B, c_mod, C, reuse_a, reuse_b)
  return __builtin_amdgcn_wmma_f32_16x16x32_bf16(false, a, false, b, (short)0, c,
                                                 false, false);
}

// --- fragment loaders -------------------------------------------------------
// A fragment (16x32, M x K), 16-bit layout: lane half h holds
//   K = h*16 .. h*16+15 (contiguous)
static __device__ inline bf16x16 loadA_f32(const float* __restrict__ base, int ld,
                                           int r0, int k0, int lane) {
  int l = lane & 15, h = (lane >> 4) & 1;
  const float* p = base + (size_t)(r0 + l) * ld + k0 + h * 16;
  bf16x16 f;
#pragma unroll
  for (int i = 0; i < 8; ++i) f[i] = (__bf16)p[i];
#pragma unroll
  for (int i = 0; i < 8; ++i) f[8 + i] = (__bf16)p[8 + i];
  return f;
}

// B fragment (32x16, K x N) supplied from a row-major [N, K] source (NT gemm):
// lane half h holds K = h*16 .. h*16+15 (contiguous), N = lane&15.
static __device__ inline bf16x16 loadB_f32(const float* __restrict__ base, int ld,
                                           int n0, int k0, int lane) {
  int l = lane & 15, h = (lane >> 4) & 1;
  const float* p = base + (size_t)(n0 + l) * ld + k0 + h * 16;
  bf16x16 f;
#pragma unroll
  for (int i = 0; i < 16; ++i) f[i] = (__bf16)p[i];
  return f;
}

// LoRA "up" fragment: U has layout [NL=2, DIM, RANK=16]; kappa = h*16+i maps to
// (layer = h, rank = i) -> each lane's 16 values are contiguous in memory.
static __device__ inline bf16x16 loadU_frag(const float* __restrict__ U, int n0,
                                            int lane) {
  int l = lane & 15, h = (lane >> 4) & 1;
  const float* p = U + (size_t)h * DIMV * RANKV + (size_t)(n0 + l) * RANKV;
  bf16x16 f;
#pragma unroll
  for (int i = 0; i < 16; ++i) f[i] = (__bf16)p[i];
  return f;
}

static __device__ inline bf16x16 loadA_bf16(const __bf16* base, int ld, int r0,
                                            int k0, int lane) {
  int l = lane & 15, h = (lane >> 4) & 1;
  const __bf16* p = base + (size_t)(r0 + l) * ld + k0 + h * 16;
  bf16x16 f;
#pragma unroll
  for (int i = 0; i < 8; ++i) f[i] = p[i];
#pragma unroll
  for (int i = 0; i < 8; ++i) f[8 + i] = p[8 + i];
  return f;
}

static __device__ inline bf16x16 loadB_bf16(const __bf16* base, int ld, int n0,
                                            int k0, int lane) {
  int l = lane & 15, h = (lane >> 4) & 1;
  const __bf16* p = base + (size_t)(n0 + l) * ld + k0 + h * 16;
  bf16x16 f;
#pragma unroll
  for (int i = 0; i < 16; ++i) f[i] = p[i];
  return f;
}

// ---------------------------------------------------------------------------
// LDS-staged NT GEMM: C[M,N] = A[M,K] @ B[N,K]^T (+bias[N]) (+ T[M,32] @ U^T).
// Requires M%128==0, N%128==0, K%32==0. Block = 128x128 tile, 8 waves, each
// wave owns a 32x64 sub-tile (2 A-frags x 4 B-frags -> 8 WMMAs / K-step).
// f32 -> bf16 conversion happens once per element per block (in staging).
// ---------------------------------------------------------------------------
__global__ __launch_bounds__(256) void gemm_nt_tiled(
    const float* __restrict__ A, int lda, const float* __restrict__ B, int ldb,
    const float* __restrict__ bias, const float* __restrict__ T,
    const float* __restrict__ U, float* __restrict__ C, int ldc, int M, int N,
    int K) {
  __shared__ __align__(32) __bf16 sA[128 * 32];
  __shared__ __align__(32) __bf16 sB[128 * 32];

  int lane = threadIdx.x & 31;
  int wid = threadIdx.x >> 5;
  int nbn = N >> 7;
  int mblk = blockIdx.x / nbn;
  int nblk = blockIdx.x % nbn;
  int wm = (wid & 3) * 32;   // wave's 32-row slice within block tile
  int wn = (wid >> 2) * 64;  // wave's 64-col slice within block tile

  // staging role: each thread converts 16 contiguous f32 -> bf16
  int srow = threadIdx.x >> 1;        // 0..127
  int scol = (threadIdx.x & 1) * 16;  // 0 or 16
  const float* Ast = A + (size_t)(mblk * 128 + srow) * lda + scol;
  const float* Bst = B + (size_t)(nblk * 128 + srow) * ldb + scol;
  __bf16* sAp = &sA[srow * 32 + scol];
  __bf16* sBp = &sB[srow * 32 + scol];

  floatx8 acc[2][4];
#pragma unroll
  for (int mi = 0; mi < 2; ++mi)
#pragma unroll
    for (int j = 0; j < 4; ++j)
      acc[mi][j] = (floatx8){0.f, 0.f, 0.f, 0.f, 0.f, 0.f, 0.f, 0.f};

  for (int k = 0; k < K; k += 32) {
    __syncthreads();  // everyone finished reading previous LDS contents
    {
      bf16x16 av, bv;
#pragma unroll
      for (int i = 0; i < 16; ++i) av[i] = (__bf16)Ast[k + i];
#pragma unroll
      for (int i = 0; i < 16; ++i) bv[i] = (__bf16)Bst[k + i];
      *(bf16x16*)sAp = av;
      *(bf16x16*)sBp = bv;
      // prefetch next K panel into cache (global_prefetch_b8)
      __builtin_prefetch(Ast + k + 32, 0, 3);
      __builtin_prefetch(Bst + k + 32, 0, 3);
    }
    __syncthreads();  // staging visible to all waves

    bf16x16 afr[2], bfr[4];
#pragma unroll
    for (int mi = 0; mi < 2; ++mi) {
      int r = wm + mi * 16;
      int l = lane & 15, h = (lane >> 4) & 1;
      const __bf16* p = &sA[(r + l) * 32 + h * 16];
#pragma unroll
      for (int i = 0; i < 8; ++i) afr[mi][i] = p[i];
#pragma unroll
      for (int i = 0; i < 8; ++i) afr[mi][8 + i] = p[8 + i];
    }
#pragma unroll
    for (int j = 0; j < 4; ++j) {
      int n = wn + j * 16;
      int l = lane & 15, h = (lane >> 4) & 1;
      const __bf16* p = &sB[(n + l) * 32 + h * 16];
#pragma unroll
      for (int i = 0; i < 16; ++i) bfr[j][i] = p[i];
    }
#pragma unroll
    for (int mi = 0; mi < 2; ++mi)
#pragma unroll
      for (int j = 0; j < 4; ++j)
        acc[mi][j] = wmma_bf16(afr[mi], bfr[j], acc[mi][j]);
  }

  if (T) {  // fused rank-32 LoRA epilogue (2 layers x rank16, COEF == 1.0)
#pragma unroll
    for (int mi = 0; mi < 2; ++mi) {
      bf16x16 tf = loadA_f32(T, 32, mblk * 128 + wm + mi * 16, 0, lane);
#pragma unroll
      for (int j = 0; j < 4; ++j) {
        bf16x16 uf = loadU_frag(U, nblk * 128 + wn + j * 16, lane);
        acc[mi][j] = wmma_bf16(tf, uf, acc[mi][j]);
      }
    }
  }

  int l = lane & 15, h = (lane >> 4) & 1;
#pragma unroll
  for (int mi = 0; mi < 2; ++mi) {
#pragma unroll
    for (int j = 0; j < 4; ++j) {
      int col = nblk * 128 + wn + j * 16 + l;
      float bv = bias ? bias[col] : 0.0f;
#pragma unroll
      for (int v = 0; v < 8; ++v) {
        int row = mblk * 128 + wm + mi * 16 + v + 8 * h;
        C[(size_t)row * ldc + col] = acc[mi][j][v] + bv;
      }
    }
  }
}

// ---------------------------------------------------------------------------
// Direct NT GEMM (for small N: rank-16/32 LoRA down-projections).
// One wave per 16x16 tile; 8 waves per block. M%16==0, N%16==0, K%32==0.
// ---------------------------------------------------------------------------
__global__ __launch_bounds__(256) void gemm_nt(
    const float* __restrict__ A, int lda, const float* __restrict__ B, int ldb,
    const float* __restrict__ bias, float* __restrict__ C, int ldc, int M,
    int N, int K) {
  int lane = threadIdx.x & 31;
  int wid = threadIdx.x >> 5;
  int ntn = N >> 4;
  int total = (M >> 4) * ntn;
  int tile = blockIdx.x * 8 + wid;
  if (tile >= total) return;
  int mt = tile / ntn;
  int nt = tile % ntn;

  floatx8 acc = {0.f, 0.f, 0.f, 0.f, 0.f, 0.f, 0.f, 0.f};
  for (int k = 0; k < K; k += 32) {
    bf16x16 af = loadA_f32(A, lda, mt * 16, k, lane);
    bf16x16 bf = loadB_f32(B, ldb, nt * 16, k, lane);
    acc = wmma_bf16(af, bf, acc);
  }
  int l = lane & 15, h = (lane >> 4) & 1;
  float bv = bias ? bias[nt * 16 + l] : 0.0f;
#pragma unroll
  for (int v = 0; v < 8; ++v) {
    int row = mt * 16 + v + 8 * h;
    C[(size_t)row * ldc + nt * 16 + l] = acc[v] + bv;
  }
}

// ---------------------------------------------------------------------------
// RMSNorm (per 128-wide head row) + RoPE + pack to bf16 [H, STOT, HD].
// Sequence concat order: encoder rows [0,512) then image rows [512,2560).
// One wave per (head, seq) row; 8 rows per block.
// ---------------------------------------------------------------------------
__global__ __launch_bounds__(256) void normrope_pack(
    const float* __restrict__ imgsrc, const float* __restrict__ encsrc,
    const float* __restrict__ w_img, const float* __restrict__ w_enc,
    const float* __restrict__ cosb, const float* __restrict__ sinb,
    int do_normrope, __bf16* __restrict__ dst) {
  int lane = threadIdx.x & 31;
  int wid = threadIdx.x >> 5;
  int row = blockIdx.x * 8 + wid;
  int h = row / STOT;
  int s = row % STOT;
  const float* src;
  const float* w;
  if (s < LTXT) {
    src = encsrc + (size_t)s * DIMV + h * HDV;
    w = w_enc;
  } else {
    src = imgsrc + (size_t)(s - LTXT) * DIMV + h * HDV;
    w = w_img;
  }
  int d0 = lane * 4;
  float x0 = src[d0 + 0], x1 = src[d0 + 1], x2 = src[d0 + 2], x3 = src[d0 + 3];
  if (do_normrope) {
    float ss = x0 * x0 + x1 * x1 + x2 * x2 + x3 * x3;
#pragma unroll
    for (int off = 1; off < 32; off <<= 1) ss += __shfl_xor(ss, off, 32);
    float r = rsqrtf(ss * (1.0f / (float)HDV) + 1e-6f);
    x0 *= r * w[d0 + 0];
    x1 *= r * w[d0 + 1];
    x2 *= r * w[d0 + 2];
    x3 *= r * w[d0 + 3];
    const float* cp = cosb + (size_t)s * HDV + d0;
    const float* sp = sinb + (size_t)s * HDV + d0;
    float y0 = x0 * cp[0] - x1 * sp[0];
    float y1 = x1 * cp[1] + x0 * sp[1];
    float y2 = x2 * cp[2] - x3 * sp[2];
    float y3 = x3 * cp[3] + x2 * sp[3];
    x0 = y0; x1 = y1; x2 = y2; x3 = y3;
  }
  __bf16* dp = dst + (size_t)h * STOT * HDV + (size_t)s * HDV + d0;
  dp[0] = (__bf16)x0;
  dp[1] = (__bf16)x1;
  dp[2] = (__bf16)x2;
  dp[3] = (__bf16)x3;
}

// ---------------------------------------------------------------------------
// Flash attention: one block per (head, 8 query-tiles); 8 waves, each owning a
// 16-query tile. K/V 32-key chunks staged cooperatively in LDS (V transposed).
// P goes through per-wave LDS scratch to re-layout C-frag -> A-frag (intra-wave
// DS ops are in-order on CDNA5, no barrier needed for that hop).
// ---------------------------------------------------------------------------
__global__ __launch_bounds__(256) void attn_fa(const __bf16* __restrict__ Qh,
                                               const __bf16* __restrict__ Kh,
                                               const __bf16* __restrict__ Vh,
                                               float* __restrict__ O) {
  __shared__ __align__(32) __bf16 sK[32 * HDV];   // 32 keys x 128, row-major
  __shared__ __align__(32) __bf16 sVt[HDV * 32];  // transposed: [d][key]
  __shared__ __align__(32) __bf16 sP[8][16 * 32]; // per-wave P scratch

  int lane = threadIdx.x & 31;
  int wid = threadIdx.x >> 5;
  int tl = threadIdx.x;
  int head = blockIdx.x;
  int q0 = (blockIdx.y * 8 + wid) * 16;

  const __bf16* Qb = Qh + (size_t)head * STOT * HDV;
  const __bf16* Kb = Kh + (size_t)head * STOT * HDV;
  const __bf16* Vb = Vh + (size_t)head * STOT * HDV;

  bf16x16 qf[4];
#pragma unroll
  for (int c = 0; c < 4; ++c) qf[c] = loadA_bf16(Qb, HDV, q0, c * 32, lane);

  floatx8 o[8];
#pragma unroll
  for (int t = 0; t < 8; ++t) o[t] = (floatx8){0.f, 0.f, 0.f, 0.f, 0.f, 0.f, 0.f, 0.f};
  float mrow[8], lrow[8], scv[8];
#pragma unroll
  for (int v = 0; v < 8; ++v) { mrow[v] = -1e30f; lrow[v] = 0.0f; }

  const float SCL = 0.08838834764831845f;  // 1/sqrt(128)
  int l = lane & 15, hh = (lane >> 4) & 1;

  for (int kc = 0; kc < STOT; kc += 32) {
    __syncthreads();
    {  // cooperative stage: K chunk row-major, V chunk transposed
      int krow = tl >> 3;
      int seg = (tl & 7) * 16;
      const uint4* src = (const uint4*)(Kb + (size_t)(kc + krow) * HDV + seg);
      uint4* dst = (uint4*)(sK + krow * HDV + seg);
      dst[0] = src[0];
      dst[1] = src[1];
      const __bf16* vs = Vb + (size_t)(kc + krow) * HDV + seg;
#pragma unroll
      for (int i = 0; i < 16; ++i) sVt[(seg + i) * 32 + krow] = vs[i];
    }
    __syncthreads();

    // scores: S = Q (16x128) x K^T, two 16-key column tiles
    floatx8 s0 = {0.f, 0.f, 0.f, 0.f, 0.f, 0.f, 0.f, 0.f};
    floatx8 s1 = {0.f, 0.f, 0.f, 0.f, 0.f, 0.f, 0.f, 0.f};
#pragma unroll
    for (int c = 0; c < 4; ++c) {
      bf16x16 k0f = loadB_bf16(sK, HDV, 0, c * 32, lane);
      s0 = wmma_bf16(qf[c], k0f, s0);
      bf16x16 k1f = loadB_bf16(sK, HDV, 16, c * 32, lane);
      s1 = wmma_bf16(qf[c], k1f, s1);
    }

    float p0[8], p1[8];
#pragma unroll
    for (int v = 0; v < 8; ++v) {
      float a = s0[v] * SCL;
      float b = s1[v] * SCL;
      float mx = fmaxf(a, b);
#pragma unroll
      for (int off = 1; off < 16; off <<= 1) mx = fmaxf(mx, __shfl_xor(mx, off, 32));
      float mn = fmaxf(mrow[v], mx);
      float sc = __expf(mrow[v] - mn);
      mrow[v] = mn;
      float e0 = __expf(a - mn);
      float e1 = __expf(b - mn);
      float rs = e0 + e1;
#pragma unroll
      for (int off = 1; off < 16; off <<= 1) rs += __shfl_xor(rs, off, 32);
      lrow[v] = lrow[v] * sc + rs;
      scv[v] = sc;
      p0[v] = e0;
      p1[v] = e1;
    }
#pragma unroll
    for (int t = 0; t < 8; ++t) {
#pragma unroll
      for (int v = 0; v < 8; ++v) o[t][v] *= scv[v];
    }

    // C-frag -> A-frag re-layout through per-wave LDS
    __bf16* pp = &sP[wid][0];
#pragma unroll
    for (int v = 0; v < 8; ++v) {
      pp[(v + 8 * hh) * 32 + l] = (__bf16)p0[v];
      pp[(v + 8 * hh) * 32 + 16 + l] = (__bf16)p1[v];
    }
    asm volatile("" ::: "memory");  // keep DS store->load program order
    bf16x16 pf = loadA_bf16(pp, 32, 0, 0, lane);
#pragma unroll
    for (int t = 0; t < 8; ++t) {
      bf16x16 vf = loadB_bf16(sVt, 32, t * 16, 0, lane);
      o[t] = wmma_bf16(pf, vf, o[t]);
    }
  }

  // finalize: divide by softmax denom, scatter to O[s, head*128 + d] (f32)
#pragma unroll
  for (int v = 0; v < 8; ++v) {
    float inv = 1.0f / lrow[v];
    int s = q0 + v + 8 * hh;
#pragma unroll
    for (int t = 0; t < 8; ++t) {
      O[(size_t)s * DIMV + head * HDV + t * 16 + l] = o[t][v] * inv;
    }
  }
}

// ---------------------------------------------------------------------------
// hs[s, d] += sum_r t[s, r] * u[d, r]   (rank-16 up-projection, in place)
// ---------------------------------------------------------------------------
__global__ __launch_bounds__(256) void rank16_update(float* __restrict__ hs,
                                                     const float* __restrict__ t,
                                                     const float* __restrict__ u,
                                                     int M) {
  int g = blockIdx.x * blockDim.x + threadIdx.x;
  const int nblk = DIMV / 8;  // 384
  int sdx = g / nblk;
  int db = g % nblk;
  if (sdx >= M) return;
  float tr[16];
#pragma unroll
  for (int r = 0; r < 16; ++r) tr[r] = t[(size_t)sdx * RANKV + r];
  int d0 = db * 8;
#pragma unroll
  for (int j = 0; j < 8; ++j) {
    const float* up = u + (size_t)(d0 + j) * RANKV;
    float acc = 0.0f;
#pragma unroll
    for (int r = 0; r < 16; ++r) acc += tr[r] * up[r];
    hs[(size_t)sdx * DIMV + d0 + j] += acc;
  }
}

// ---------------------------------------------------------------------------
extern "C" void kernel_launch(void* const* d_in, const int* in_sizes, int n_in,
                              void* d_out, int out_size, void* d_ws,
                              size_t ws_size, hipStream_t stream) {
  (void)in_sizes; (void)n_in; (void)out_size; (void)ws_size;
  const float* hs   = (const float*)d_in[0];
  const float* enc  = (const float*)d_in[1];
  const float* cosb = (const float*)d_in[2];
  const float* sinb = (const float*)d_in[3];
  const float* Wq  = (const float*)d_in[4];
  const float* Wk  = (const float*)d_in[5];
  const float* Wv  = (const float*)d_in[6];
  const float* Wo  = (const float*)d_in[7];
  const float* Waq = (const float*)d_in[8];
  const float* Wak = (const float*)d_in[9];
  const float* Wav = (const float*)d_in[10];
  const float* Wao = (const float*)d_in[11];
  const float* bq  = (const float*)d_in[12];
  const float* bk  = (const float*)d_in[13];
  const float* bv  = (const float*)d_in[14];
  const float* bo  = (const float*)d_in[15];
  const float* baq = (const float*)d_in[16];
  const float* bak = (const float*)d_in[17];
  const float* bav = (const float*)d_in[18];
  const float* bao = (const float*)d_in[19];
  const float* nq  = (const float*)d_in[20];
  const float* nk  = (const float*)d_in[21];
  const float* naq = (const float*)d_in[22];
  const float* nak = (const float*)d_in[23];
  const float* qd  = (const float*)d_in[24];
  const float* kd  = (const float*)d_in[25];
  const float* vd  = (const float*)d_in[26];
  const float* pd  = (const float*)d_in[27];
  const float* qu  = (const float*)d_in[28];
  const float* ku  = (const float*)d_in[29];
  const float* vu  = (const float*)d_in[30];
  const float* pu  = (const float*)d_in[31];

  const size_t IMG = (size_t)LIMG * DIMV;           // image activation elems
  const size_t TXT = (size_t)LTXT * DIMV;           // encoder activation elems
  const size_t HSD = (size_t)HEADSV * STOT * HDV;   // head-major elems

  float* wsf   = (float*)d_ws;
  float* qbuf  = wsf;              // [2048,3072] f32
  float* kbuf  = qbuf + IMG;
  float* vbuf  = kbuf + IMG;
  float* eqb   = vbuf + IMG;       // [512,3072] f32
  float* ekb   = eqb + TXT;
  float* evb   = ekb + TXT;
  float* Tq    = evb + TXT;        // [2048,32] f32
  float* Tk    = Tq + (size_t)LIMG * 32;
  float* Tv    = Tk + (size_t)LIMG * 32;
  float* tpost = Tv + (size_t)LIMG * 32;  // [2048,16] f32
  __bf16* Qh   = (__bf16*)(tpost + (size_t)LIMG * RANKV);
  __bf16* Kh   = Qh + HSD;
  __bf16* Vh   = Kh + HSD;
  float* Oattn = qbuf;  // alias: q/k f32 buffers are dead by attention time

  float* out_hs  = (float*)d_out;        // [2048,3072]
  float* out_enc = out_hs + IMG;         // [512,3072]

  auto sblocks = [](int M, int N) { return ((M / 16) * (N / 16) + 7) / 8; };
  auto tblocks = [](int M, int N) { return (M / 128) * (N / 128); };

  // 1) LoRA down-projections T = hs @ [d0;d1]^T  (qd/kd/vd are [2,16,3072])
  gemm_nt<<<sblocks(LIMG, 32), 256, 0, stream>>>(hs, DIMV, qd, DIMV, nullptr,
                                                 Tq, 32, LIMG, 32, DIMV);
  gemm_nt<<<sblocks(LIMG, 32), 256, 0, stream>>>(hs, DIMV, kd, DIMV, nullptr,
                                                 Tk, 32, LIMG, 32, DIMV);
  gemm_nt<<<sblocks(LIMG, 32), 256, 0, stream>>>(hs, DIMV, vd, DIMV, nullptr,
                                                 Tv, 32, LIMG, 32, DIMV);

  // 2) image QKV with fused bias + LoRA epilogue (LDS-staged tiled GEMM)
  gemm_nt_tiled<<<tblocks(LIMG, DIMV), 256, 0, stream>>>(
      hs, DIMV, Wq, DIMV, bq, Tq, qu, qbuf, DIMV, LIMG, DIMV, DIMV);
  gemm_nt_tiled<<<tblocks(LIMG, DIMV), 256, 0, stream>>>(
      hs, DIMV, Wk, DIMV, bk, Tk, ku, kbuf, DIMV, LIMG, DIMV, DIMV);
  gemm_nt_tiled<<<tblocks(LIMG, DIMV), 256, 0, stream>>>(
      hs, DIMV, Wv, DIMV, bv, Tv, vu, vbuf, DIMV, LIMG, DIMV, DIMV);

  // 3) encoder QKV
  gemm_nt_tiled<<<tblocks(LTXT, DIMV), 256, 0, stream>>>(
      enc, DIMV, Waq, DIMV, baq, nullptr, nullptr, eqb, DIMV, LTXT, DIMV, DIMV);
  gemm_nt_tiled<<<tblocks(LTXT, DIMV), 256, 0, stream>>>(
      enc, DIMV, Wak, DIMV, bak, nullptr, nullptr, ekb, DIMV, LTXT, DIMV, DIMV);
  gemm_nt_tiled<<<tblocks(LTXT, DIMV), 256, 0, stream>>>(
      enc, DIMV, Wav, DIMV, bav, nullptr, nullptr, evb, DIMV, LTXT, DIMV, DIMV);

  // 4) RMSNorm + RoPE + pack head-major bf16
  int nrBlocks = (HEADSV * STOT) / 8;
  normrope_pack<<<nrBlocks, 256, 0, stream>>>(qbuf, eqb, nq, naq, cosb, sinb, 1, Qh);
  normrope_pack<<<nrBlocks, 256, 0, stream>>>(kbuf, ekb, nk, nak, cosb, sinb, 1, Kh);
  normrope_pack<<<nrBlocks, 256, 0, stream>>>(vbuf, evb, nullptr, nullptr, nullptr,
                                              nullptr, 0, Vh);

  // 5) flash attention (24 heads x 160 query tiles, 8 tiles per block)
  attn_fa<<<dim3(HEADSV, STOT / 16 / 8), 256, 0, stream>>>(Qh, Kh, Vh, Oattn);

  // 6) image output projection -> d_out, then two sequential rank-16 updates
  gemm_nt_tiled<<<tblocks(LIMG, DIMV), 256, 0, stream>>>(
      Oattn + TXT, DIMV, Wo, DIMV, bo, nullptr, nullptr, out_hs, DIMV, LIMG,
      DIMV, DIMV);
  for (int n = 0; n < NLV; ++n) {
    gemm_nt<<<sblocks(LIMG, RANKV), 256, 0, stream>>>(
        out_hs, DIMV, pd + (size_t)n * RANKV * DIMV, DIMV, nullptr, tpost,
        RANKV, LIMG, RANKV, DIMV);
    rank16_update<<<((LIMG * (DIMV / 8)) + 255) / 256, 256, 0, stream>>>(
        out_hs, tpost, pu + (size_t)n * DIMV * RANKV, LIMG);
  }

  // 7) encoder output projection
  gemm_nt_tiled<<<tblocks(LTXT, DIMV), 256, 0, stream>>>(
      Oattn, DIMV, Wao, DIMV, bao, nullptr, nullptr, out_enc, DIMV, LTXT, DIMV,
      DIMV);
}
